// MS_BACL_55765855371540
// MI455X (gfx1250) — matrix operand order repaired and checked by reading, compiled, not run on hardware
//
#include <hip/hip_runtime.h>
#include <hip/hip_bf16.h>
#include <stdint.h>

typedef __attribute__((ext_vector_type(16))) _Float16 v16h;
typedef __attribute__((ext_vector_type(8)))  _Float16 v8h;
typedef __attribute__((ext_vector_type(8)))  float    v8f;

#define BN 64
#define BM 64
#define BK 32
#define AS_STRIDE 40   // 80B rows: 16B-aligned, conflict-free fragment reads
#define BS_STRIDE 40
#define GNUM 256

// ---------------- utility kernels ----------------

__global__ void zero_f32_kernel(float* __restrict__ buf, long n) {
  long i = (long)blockIdx.x * blockDim.x + threadIdx.x;
  if (i < n) buf[i] = 0.0f;
}

__global__ void copy_f32_kernel(float* __restrict__ dst, const float* __restrict__ src, long n) {
  long i = (long)blockIdx.x * blockDim.x + threadIdx.x;
  if (i < n) dst[i] = src[i];
}

// agg[dst] += h[src] over all edges, feature-parallel
__global__ void edge_scatter_add_kernel(const float* __restrict__ h,
                                        const int* __restrict__ ei,
                                        float* __restrict__ agg, int E, int F) {
  long idx = (long)blockIdx.x * blockDim.x + threadIdx.x;
  long total = (long)E * F;
  if (idx >= total) return;
  int e = (int)(idx / F);
  int f = (int)(idx - (long)e * F);
  int s = ei[e];
  int d = ei[E + e];
  atomicAdd(&agg[(long)d * F + f], h[(long)s * F + f]);
}

__global__ void count_nodes_kernel(const int* __restrict__ seg, int* __restrict__ cnt, int n) {
  int i = blockIdx.x * blockDim.x + threadIdx.x;
  if (i < n) atomicAdd(&cnt[seg[i]], 1);
}

// p layout per graph: [0..M) = max, [M..2M) = sum -> becomes mean
__global__ void pool_finalize_kernel(float* __restrict__ p, const int* __restrict__ cnt, int M) {
  long idx = (long)blockIdx.x * blockDim.x + threadIdx.x;
  long total = (long)GNUM * M;
  if (idx >= total) return;
  int g = (int)(idx / M);
  int j = (int)(idx - (long)g * M);
  float c = (float)cnt[g];
  p[(long)g * 2 * M + M + j] /= fmaxf(c, 1.0f);
}

// final layer: z[g] = dot(U[g,:256], w) + b
__global__ void dot_out_kernel(const float* __restrict__ U, const float* __restrict__ w,
                               const float* __restrict__ b, float* __restrict__ z, int K) {
  __shared__ float red[256];
  int g = blockIdx.x;
  int t = threadIdx.x;
  float v = (t < K) ? U[(long)g * K + t] * w[t] : 0.0f;
  red[t] = v;
  __syncthreads();
  for (int s = 128; s > 0; s >>= 1) {
    if (t < s) red[t] += red[t + s];
    __syncthreads();
  }
  if (t == 0) z[g] = red[0] + b[0];
}

// ---------------- WMMA GEMM: C = act(A[N,K] @ W[K,M] + bias) ----------------
// 256 threads = 8 wave32; block tile 64x64; each wave: 16 rows x 32 cols
// (2x v_wmma_f32_16x16x32_f16 per K-chunk). Optional fused pooling epilogue:
// atomic max/sum per graph (never materializes C).
__global__ __launch_bounds__(256)
void gemm_wmma_kernel(const float* __restrict__ A, const float* __restrict__ W,
                      const float* __restrict__ bias, float* __restrict__ C,
                      int N, int K, int M, int relu,
                      const int* __restrict__ seg, float* __restrict__ pool, int do_pool) {
  __shared__ _Float16 As[BN * AS_STRIDE];
  __shared__ _Float16 Bs[BM * BS_STRIDE];

  const int tid  = threadIdx.x;
  const int wave = tid >> 5;
  const int lane = tid & 31;
  const int wr   = wave & 3;   // row sub-tile (16 rows)
  const int wc   = wave >> 2;  // col sub-tile (32 cols)
  const int row0 = blockIdx.x * BN;
  const int col0 = blockIdx.y * BM;
  const int l16  = lane & 15;
  const int lhi  = lane >> 4;

  // staging coordinates (hoisted out of K loop)
  const int ar  = tid >> 2;             // A: row in tile 0..63
  const int ak0 = (tid & 3) * 8;        // A: k start 0,8,16,24
  const int agr = row0 + ar;            // A: global row
  const bool arow_ok = (agr < N);
  const int bk  = tid >> 3;             // B: k in tile 0..31
  const int bn0 = (tid & 7) * 8;        // B: n start 0..56

  v8f acc0 = {};
  v8f acc1 = {};

  for (int kk = 0; kk < K; kk += BK) {
    // ---- stage A tile (64 rows x 32 k) as f16, zero-padded ----
    // unpredicated loads via clamped addresses -> single load clause
    {
      float f[8];
      bool ok[8];
      #pragma unroll
      for (int i = 0; i < 8; ++i) {
        int gk = kk + ak0 + i;
        ok[i] = arow_ok && (gk < K);
        long addr = ok[i] ? ((long)agr * K + gk) : 0;
        f[i] = A[addr];
      }
      v8h hv;
      #pragma unroll
      for (int i = 0; i < 8; ++i) hv[i] = (_Float16)(ok[i] ? f[i] : 0.0f);
      *(v8h*)&As[ar * AS_STRIDE + ak0] = hv;   // 16B-aligned wide DS store
    }
    // ---- stage B tile transposed Bs[n][k]: k-pairs contiguous 32-bit words ----
    {
      int gk = kk + bk;
      bool krow_ok = (gk < K);
      float f[8];
      bool ok[8];
      #pragma unroll
      for (int i = 0; i < 8; ++i) {
        int gc = col0 + bn0 + i;
        ok[i] = krow_ok && (gc < M);
        long addr = ok[i] ? ((long)gk * M + gc) : 0;
        f[i] = W[addr];
      }
      _Float16 hv[8];
      #pragma unroll
      for (int i = 0; i < 8; ++i) hv[i] = (_Float16)(ok[i] ? f[i] : 0.0f);
      #pragma unroll
      for (int i = 0; i < 8; ++i) Bs[(bn0 + i) * BS_STRIDE + bk] = hv[i];
    }
    __syncthreads();

    // build fragments: lane half h holds k = 16h..16h+15 (contiguous order, same for A and B)
    union { v16h h; unsigned u[8]; } fa, fb0, fb1;
    #pragma unroll
    for (int v = 0; v < 8; ++v) {
      int m  = wr * 16 + l16;
      int ka = lhi * 16 + 2 * v;
      fa.u[v] = *(const unsigned*)&As[m * AS_STRIDE + ka];
      int kb  = lhi * 16 + 2 * v;
      int nb0 = wc * 32 + l16;
      fb0.u[v] = *(const unsigned*)&Bs[nb0 * BS_STRIDE + kb];
      fb1.u[v] = *(const unsigned*)&Bs[(nb0 + 16) * BS_STRIDE + kb];
    }
    acc0 = __builtin_amdgcn_wmma_f32_16x16x32_f16(false, fa.h, false, fb0.h,
                                                  (short)0, acc0, false, false);
    acc1 = __builtin_amdgcn_wmma_f32_16x16x32_f16(false, fa.h, false, fb1.h,
                                                  (short)0, acc1, false, false);
    __syncthreads();
  }

  // epilogue: bias + relu + (store | fused pool atomics)
  #pragma unroll
  for (int v = 0; v < 8; ++v) {
    int m = row0 + wr * 16 + lhi * 8 + v;
    if (m >= N) continue;
    int c0 = col0 + wc * 32 + l16;
    int c1 = c0 + 16;
    float x0 = acc0[v];
    float x1 = acc1[v];
    if (c0 < M) {
      float val = x0 + bias[c0];
      if (relu) val = fmaxf(val, 0.0f);
      if (do_pool) {
        int g = seg[m];
        atomicAdd(&pool[(long)g * (2 * M) + M + c0], val);
        atomicMax((unsigned*)&pool[(long)g * (2 * M) + c0], __float_as_uint(val));
      } else {
        C[(long)m * M + c0] = val;
      }
    }
    if (c1 < M) {
      float val = x1 + bias[c1];
      if (relu) val = fmaxf(val, 0.0f);
      if (do_pool) {
        int g = seg[m];
        atomicAdd(&pool[(long)g * (2 * M) + M + c1], val);
        atomicMax((unsigned*)&pool[(long)g * (2 * M) + c1], __float_as_uint(val));
      } else {
        C[(long)m * M + c1] = val;
      }
    }
  }
}

// ---------------- host-side launch ----------------

static inline int cdiv(long a, long b) { return (int)((a + b - 1) / b); }

static void launch_gemm(const float* A, const float* W, const float* bias, float* C,
                        int N, int K, int M, int relu,
                        const int* seg, float* pool, int do_pool, hipStream_t stream) {
  dim3 grid(cdiv(N, BN), cdiv(M, BM));
  gemm_wmma_kernel<<<grid, 256, 0, stream>>>(A, W, bias, C, N, K, M, relu, seg, pool, do_pool);
}

extern "C" void kernel_launch(void* const* d_in, const int* in_sizes, int n_in,
                              void* d_out, int out_size, void* d_ws, size_t ws_size,
                              hipStream_t stream) {
  const int N1 = 50000, E1 = 400000;
  const int N2 = 50000, E2 = 400000;
  const int F1 = 93, F2 = 43;
  const int D1 = F1 * 10;   // 930
  const int D2 = F2 * 10;   // 430

  const float* x     = (const float*)d_in[1];
  const int*   ei    = (const int*)d_in[2];
  const int*   batch = (const int*)d_in[3];
  const float* aIn   = (const float*)d_in[4];
  const int*   edge  = (const int*)d_in[5];
  const int*   cseg  = (const int*)d_in[6];
  const float* W1 = (const float*)d_in[7],  *b1 = (const float*)d_in[8];
  const float* W2 = (const float*)d_in[9],  *b2 = (const float*)d_in[10];
  const float* W3 = (const float*)d_in[11], *b3 = (const float*)d_in[12];
  const float* W4 = (const float*)d_in[13], *b4 = (const float*)d_in[14];
  const float* fcg_w1  = (const float*)d_in[15], *fcg_b1  = (const float*)d_in[16];
  const float* fcg_w2  = (const float*)d_in[17], *fcg_b2  = (const float*)d_in[18];
  const float* fcg1_w1 = (const float*)d_in[19], *fcg1_b1 = (const float*)d_in[20];
  const float* fcg1_w2 = (const float*)d_in[21], *fcg1_b2 = (const float*)d_in[22];
  const float* ff_w1  = (const float*)d_in[23], *ff_b1  = (const float*)d_in[24];
  const float* ff_w2  = (const float*)d_in[25], *ff_b2  = (const float*)d_in[26];
  const float* ff1_w1 = (const float*)d_in[27], *ff1_b1 = (const float*)d_in[28];
  const float* ff1_w2 = (const float*)d_in[29], *ff1_b2 = (const float*)d_in[30];

  float* out = (float*)d_out;
  float* z   = out;                          // [256,1]
  float* xg  = out + GNUM;                   // [256,512]
  float* xg1 = out + GNUM + GNUM * 512;      // [256,512]
  float* z1  = out + GNUM + 2 * GNUM * 512;  // [256,1]

  // workspace carve-out (256B aligned chunks)
  char* wsp = (char*)d_ws;
  auto carve = [&](size_t bytes) -> void* {
    void* p = (void*)wsp;
    wsp += (bytes + 255) & ~(size_t)255;
    return p;
  };
  float* agg1 = (float*)carve((size_t)N1 * F1 * 4);
  float* h1   = (float*)carve((size_t)N1 * F1 * 4);
  float* p    = (float*)carve((size_t)GNUM * 2 * D1 * 4);
  float* t    = (float*)carve((size_t)GNUM * 1024 * 4);
  float* u    = (float*)carve((size_t)GNUM * 256 * 4);
  int*   cnt1 = (int*)  carve((size_t)GNUM * 4);
  float* aggB = (float*)carve((size_t)N2 * F2 * 4);
  float* h1b  = (float*)carve((size_t)N2 * F2 * 4);
  float* p1   = (float*)carve((size_t)GNUM * 2 * D2 * 4);
  float* t1   = (float*)carve((size_t)GNUM * 1024 * 4);
  float* u1   = (float*)carve((size_t)GNUM * 256 * 4);
  int*   cnt2 = (int*)  carve((size_t)GNUM * 4);

  const int TB = 256;
  long n;

  // ======================= Branch 1 (93-dim) =======================
  // GIN1: agg = x + scatter_add(x)
  n = (long)N1 * F1;
  copy_f32_kernel<<<cdiv(n, TB), TB, 0, stream>>>(agg1, x, n);
  n = (long)E1 * F1;
  edge_scatter_add_kernel<<<cdiv(n, TB), TB, 0, stream>>>(x, ei, agg1, E1, F1);
  launch_gemm(agg1, W1, b1, h1, N1, F1, F1, /*relu*/1, nullptr, nullptr, 0, stream);

  // GIN2 aggregation (reuse agg1 buffer)
  n = (long)N1 * F1;
  copy_f32_kernel<<<cdiv(n, TB), TB, 0, stream>>>(agg1, h1, n);
  n = (long)E1 * F1;
  edge_scatter_add_kernel<<<cdiv(n, TB), TB, 0, stream>>>(h1, ei, agg1, E1, F1);

  // pooling prep: zero pool buffer + counts
  n = (long)GNUM * 2 * D1;
  zero_f32_kernel<<<cdiv(n, TB), TB, 0, stream>>>(p, n);
  zero_f32_kernel<<<1, TB, 0, stream>>>((float*)cnt1, GNUM);
  count_nodes_kernel<<<cdiv(N1, TB), TB, 0, stream>>>(batch, cnt1, N1);

  // GIN2 linear with fused relu + max/sum pooling (h2 never materialized)
  launch_gemm(agg1, W2, b2, nullptr, N1, F1, D1, /*relu*/1, batch, p, /*pool*/1, stream);
  n = (long)GNUM * D1;
  pool_finalize_kernel<<<cdiv(n, TB), TB, 0, stream>>>(p, cnt1, D1);

  // MLP head
  launch_gemm(p,  fcg_w1, fcg_b1, t,  GNUM, 2 * D1, 1024, 1, nullptr, nullptr, 0, stream);
  launch_gemm(t,  fcg_w2, fcg_b2, xg, GNUM, 1024,   512,  0, nullptr, nullptr, 0, stream);
  launch_gemm(xg, ff_w1,  ff_b1,  u,  GNUM, 512,    256,  1, nullptr, nullptr, 0, stream);
  dot_out_kernel<<<GNUM, 256, 0, stream>>>(u, ff_w2, ff_b2, z, 256);

  // ======================= Branch 2 (43-dim) =======================
  n = (long)N2 * F2;
  copy_f32_kernel<<<cdiv(n, TB), TB, 0, stream>>>(aggB, aIn, n);
  n = (long)E2 * F2;
  edge_scatter_add_kernel<<<cdiv(n, TB), TB, 0, stream>>>(aIn, edge, aggB, E2, F2);
  launch_gemm(aggB, W3, b3, h1b, N2, F2, F2, 1, nullptr, nullptr, 0, stream);

  n = (long)N2 * F2;
  copy_f32_kernel<<<cdiv(n, TB), TB, 0, stream>>>(aggB, h1b, n);
  n = (long)E2 * F2;
  edge_scatter_add_kernel<<<cdiv(n, TB), TB, 0, stream>>>(h1b, edge, aggB, E2, F2);

  n = (long)GNUM * 2 * D2;
  zero_f32_kernel<<<cdiv(n, TB), TB, 0, stream>>>(p1, n);
  zero_f32_kernel<<<1, TB, 0, stream>>>((float*)cnt2, GNUM);
  count_nodes_kernel<<<cdiv(N2, TB), TB, 0, stream>>>(cseg, cnt2, N2);

  launch_gemm(aggB, W4, b4, nullptr, N2, F2, D2, 1, cseg, p1, 1, stream);
  n = (long)GNUM * D2;
  pool_finalize_kernel<<<cdiv(n, TB), TB, 0, stream>>>(p1, cnt2, D2);

  launch_gemm(p1,  fcg1_w1, fcg1_b1, t1,  GNUM, 2 * D2, 1024, 1, nullptr, nullptr, 0, stream);
  launch_gemm(t1,  fcg1_w2, fcg1_b2, xg1, GNUM, 1024,   512,  0, nullptr, nullptr, 0, stream);
  launch_gemm(xg1, ff1_w1,  ff1_b1,  u1,  GNUM, 512,    256,  1, nullptr, nullptr, 0, stream);
  dot_out_kernel<<<GNUM, 256, 0, stream>>>(u1, ff1_w2, ff1_b2, z1, 256);
}
